// LSTM_31714038513774
// MI455X (gfx1250) — matrix operand order repaired and checked by reading, not compiled, not run on hardware
//
#include <hip/hip_runtime.h>

#define NB 4096   // batch
#define NT 15     // seq len
#define NE 300    // embed dim
#define NEP 320   // padded embed dim (multiple of 32)
#define NH 512    // hidden
#define NG 1536   // 3*hidden (gates)
#define ND 1024   // 2*hidden (TextSA dim)

typedef __attribute__((ext_vector_type(16))) __bf16 v16bf;
typedef __attribute__((ext_vector_type(8)))  float  v8f;
typedef __attribute__((ext_vector_type(4)))  unsigned int v4u;

__device__ __forceinline__ unsigned short f32_to_bf16(float f) {
  unsigned int u = __float_as_uint(f);
  u += 0x7fffu + ((u >> 16) & 1u);   // round-to-nearest-even
  return (unsigned short)(u >> 16);
}

union FragU { v16bf v; v4u q[2]; };

// A-matrix 16x32 bf16 fragment: lane half h -> k = {h*8+0..7, 16+h*8+0..7}
__device__ __forceinline__ v16bf load_frag_a(const unsigned short* __restrict__ row, int k, int half) {
  FragU f;
  f.q[0] = *reinterpret_cast<const v4u*>(row + k + half * 8);
  f.q[1] = *reinterpret_cast<const v4u*>(row + k + 16 + half * 8);
  return f.v;
}
// B-matrix 32x16 bf16 fragment, column-major source (row n of W = column n of B):
// lane half h -> k = {h*8+0..7, 16+h*8+0..7}
__device__ __forceinline__ v16bf load_frag_b(const unsigned short* __restrict__ row, int k, int half) {
  FragU f;
  const unsigned short* p = row + k + half * 8;
  f.q[0] = *reinterpret_cast<const v4u*>(p);
  f.q[1] = *reinterpret_cast<const v4u*>(p + 16);
  return f.v;
}

#define WMMA(a, b, c) __builtin_amdgcn_wmma_f32_16x16x32_bf16(false, (a), false, (b), (short)0, (c), false, false)

// ---------------- generic bf16 WMMA GEMM: C[M,N] = A[M,K] * W[N,K]^T (+bias, tanh) ----
// grid = (N/64, M/64), block = 128 (4 waves, each a 16x64 tile)
__global__ __launch_bounds__(128)
void wmma_gemm_bf16_kernel(const unsigned short* __restrict__ A, int lda,
                           const unsigned short* __restrict__ W, int ldw,
                           float* __restrict__ C, int ldc, int K,
                           const float* __restrict__ bias, int act_tanh) {
  const int lane = threadIdx.x & 31;
  const int wave = threadIdx.x >> 5;
  const int half = lane >> 4;
  const int lm   = lane & 15;
  const int nTile = blockIdx.x * 64;
  const int mTile = blockIdx.y * 64 + wave * 16;
  const unsigned short* aRow = A + (size_t)(mTile + lm) * lda;
  const unsigned short* wRow = W + (size_t)(nTile + lm) * ldw;
  const size_t wStep = (size_t)16 * ldw;
  v8f c0 = {}, c1 = {}, c2 = {}, c3 = {};
  for (int k = 0; k < K; k += 32) {
    v16bf a  = load_frag_a(aRow, k, half);
    v16bf b0 = load_frag_b(wRow,             k, half);
    v16bf b1 = load_frag_b(wRow +     wStep, k, half);
    v16bf b2 = load_frag_b(wRow + 2 * wStep, k, half);
    v16bf b3 = load_frag_b(wRow + 3 * wStep, k, half);
    c0 = WMMA(a, b0, c0);
    c1 = WMMA(a, b1, c1);
    c2 = WMMA(a, b2, c2);
    c3 = WMMA(a, b3, c3);
  }
  float bv0 = 0.f, bv1 = 0.f, bv2 = 0.f, bv3 = 0.f;
  if (bias) {
    bv0 = bias[nTile + lm];      bv1 = bias[nTile + 16 + lm];
    bv2 = bias[nTile + 32 + lm]; bv3 = bias[nTile + 48 + lm];
  }
#pragma unroll
  for (int r = 0; r < 8; ++r) {
    float v0 = c0[r] + bv0, v1 = c1[r] + bv1, v2 = c2[r] + bv2, v3 = c3[r] + bv3;
    if (act_tanh) { v0 = tanhf(v0); v1 = tanhf(v1); v2 = tanhf(v2); v3 = tanhf(v3); }
    float* cr = C + (size_t)(mTile + half * 8 + r) * ldc + nTile + lm;
    cr[0] = v0; cr[16] = v1; cr[32] = v2; cr[48] = v3;
  }
}

// ---------------- fused attention-weights kernel --------------------------------------
// weights[m] += sum_n tanh(yw[m]·w_local[n] + b_local[n]) * g_emb[m/T][n] * w_common[n]
// grid = (ND/64, (NB*NT)/64), block = 128. weights pre-initialized to b_common.
__global__ __launch_bounds__(128)
void attn_weights_kernel(const unsigned short* __restrict__ ywbf,
                         const unsigned short* __restrict__ wl,
                         const float* __restrict__ b_local,
                         const float* __restrict__ g_emb,
                         const float* __restrict__ w_common,
                         float* __restrict__ wts) {
  const int lane = threadIdx.x & 31;
  const int wave = threadIdx.x >> 5;
  const int half = lane >> 4;
  const int lm   = lane & 15;
  const int nTile = blockIdx.x * 64;
  const int mTile = blockIdx.y * 64 + wave * 16;
  const unsigned short* aRow = ywbf + (size_t)(mTile + lm) * ND;
  const unsigned short* wRow = wl + (size_t)(nTile + lm) * ND;
  const size_t wStep = (size_t)16 * ND;
  v8f c0 = {}, c1 = {}, c2 = {}, c3 = {};
  for (int k = 0; k < ND; k += 32) {
    v16bf a  = load_frag_a(aRow, k, half);
    v16bf b0 = load_frag_b(wRow,             k, half);
    v16bf b1 = load_frag_b(wRow +     wStep, k, half);
    v16bf b2 = load_frag_b(wRow + 2 * wStep, k, half);
    v16bf b3 = load_frag_b(wRow + 3 * wStep, k, half);
    c0 = WMMA(a, b0, c0);
    c1 = WMMA(a, b1, c1);
    c2 = WMMA(a, b2, c2);
    c3 = WMMA(a, b3, c3);
  }
  const float bl0 = b_local[nTile + lm],      bl1 = b_local[nTile + 16 + lm];
  const float bl2 = b_local[nTile + 32 + lm], bl3 = b_local[nTile + 48 + lm];
  const float wc0 = w_common[nTile + lm],      wc1 = w_common[nTile + 16 + lm];
  const float wc2 = w_common[nTile + 32 + lm], wc3 = w_common[nTile + 48 + lm];
  float part[8];
#pragma unroll
  for (int r = 0; r < 8; ++r) {
    const int m = mTile + half * 8 + r;
    const int bb = m / NT;
    const float* ge = g_emb + (size_t)bb * ND + nTile;
    part[r] = tanhf(c0[r] + bl0) * ge[lm]      * wc0
            + tanhf(c1[r] + bl1) * ge[16 + lm] * wc1
            + tanhf(c2[r] + bl2) * ge[32 + lm] * wc2
            + tanhf(c3[r] + bl3) * ge[48 + lm] * wc3;
  }
  // reduce across the 16 columns held by each 16-lane half (wave32)
#pragma unroll
  for (int off = 1; off < 16; off <<= 1)
#pragma unroll
    for (int r = 0; r < 8; ++r) part[r] += __shfl_xor(part[r], off, 32);
  if (lm == 0) {
#pragma unroll
    for (int r = 0; r < 8; ++r) atomicAdd(&wts[mTile + half * 8 + r], part[r]);
  }
}

// ---------------- pointwise / data movement kernels -----------------------------------
__global__ void gather_embed_kernel(const int* __restrict__ inds, const float* __restrict__ emb,
                                    unsigned short* __restrict__ x_bf) {
  size_t idx = (size_t)blockIdx.x * blockDim.x + threadIdx.x;
  if (idx >= (size_t)NB * NT * NEP) return;
  int k = (int)(idx % NEP);
  size_t bt = idx / NEP;
  int tok = inds[bt];
  float v = (k < NE) ? emb[(size_t)tok * NE + k] : 0.f;
  x_bf[idx] = f32_to_bf16(v);
}

__global__ void convert_pad_kernel(const float* __restrict__ src, unsigned short* __restrict__ dst,
                                   int rows, int srcK, int dstK) {
  size_t idx = (size_t)blockIdx.x * blockDim.x + threadIdx.x;
  if (idx >= (size_t)rows * dstK) return;
  int k = (int)(idx % dstK);
  size_t r = idx / dstK;
  float v = (k < srcK) ? src[r * srcK + k] : 0.f;
  dst[idx] = f32_to_bf16(v);
}

__global__ void fill_f32_kernel(float* __restrict__ p, float v, size_t n) {
  size_t i = (size_t)blockIdx.x * blockDim.x + threadIdx.x;
  if (i < n) p[i] = v;
}
__global__ void fill_u16_kernel(unsigned short* __restrict__ p, unsigned short v, size_t n) {
  size_t i = (size_t)blockIdx.x * blockDim.x + threadIdx.x;
  if (i < n) p[i] = v;
}
__global__ void fill_from_ptr_kernel(float* __restrict__ p, const float* __restrict__ v, size_t n) {
  size_t i = (size_t)blockIdx.x * blockDim.x + threadIdx.x;
  if (i < n) p[i] = v[0];
}

// GRU gate math: r=sig(ir+hr), z=sig(iz+hz), n=tanh(in + r*hn), h'=(1-z)n+z h
__global__ void gru_gate_kernel(const float* __restrict__ gi, const float* __restrict__ gh,
                                const float* __restrict__ b_ih, const float* __restrict__ b_hh,
                                float* __restrict__ h_f32, unsigned short* __restrict__ h_bf,
                                float* __restrict__ y_f32, unsigned short* __restrict__ y_bf,
                                int t, int dirOff) {
  size_t idx = (size_t)blockIdx.x * blockDim.x + threadIdx.x;
  if (idx >= (size_t)NB * NH) return;
  int b = (int)(idx >> 9);
  int j = (int)(idx & (NH - 1));
  size_t base = (size_t)b * NG;
  float ir = gi[base + j]            + b_ih[j];
  float iz = gi[base + NH + j]       + b_ih[NH + j];
  float in_= gi[base + 2 * NH + j]   + b_ih[2 * NH + j];
  float hr = gh[base + j]            + b_hh[j];
  float hz = gh[base + NH + j]       + b_hh[NH + j];
  float hn = gh[base + 2 * NH + j]   + b_hh[2 * NH + j];
  float r = 1.f / (1.f + expf(-(ir + hr)));
  float z = 1.f / (1.f + expf(-(iz + hz)));
  float n = tanhf(in_ + r * hn);
  float h = h_f32[idx];
  float hnew = (1.f - z) * n + z * h;
  h_f32[idx] = hnew;
  h_bf[idx] = f32_to_bf16(hnew);
  size_t yo = ((size_t)b * NT + t) * ND + dirOff + j;
  y_f32[yo] = hnew;
  y_bf[yo] = f32_to_bf16(hnew);
}

__global__ void maxpool_kernel(const int* __restrict__ inds, const float* __restrict__ yw,
                               float* __restrict__ yp_f, unsigned short* __restrict__ yp_bf) {
  size_t idx = (size_t)blockIdx.x * blockDim.x + threadIdx.x;
  if (idx >= (size_t)NB * ND) return;
  int b = (int)(idx >> 10);
  int d = (int)(idx & (ND - 1));
  float m = -3.0e38f;
  for (int t = 0; t < NT; ++t)
    if (inds[b * NT + t] != 0)
      m = fmaxf(m, yw[((size_t)b * NT + t) * ND + d]);
  yp_f[idx] = m;
  yp_bf[idx] = f32_to_bf16(m);
}

__global__ void softmax_wsum_kernel(const float* __restrict__ wts, const float* __restrict__ yw,
                                    float* __restrict__ out) {
  const int b = blockIdx.x;
  __shared__ float p[NT];
  if (threadIdx.x == 0) {
    float mx = -3.0e38f;
    for (int t = 0; t < NT; ++t) mx = fmaxf(mx, wts[b * NT + t]);
    float e[NT]; float s = 0.f;
    for (int t = 0; t < NT; ++t) { e[t] = expf(wts[b * NT + t] - mx); s += e[t]; }
    float inv = 1.f / s;
    for (int t = 0; t < NT; ++t) p[t] = e[t] * inv;
  }
  __syncthreads();
  for (int d = threadIdx.x; d < ND; d += blockDim.x) {
    float acc = 0.f;
    for (int t = 0; t < NT; ++t) acc += p[t] * yw[((size_t)b * NT + t) * ND + d];
    out[(size_t)b * ND + d] = acc;
  }
}

// --------------------------------------------------------------------------------------
extern "C" void kernel_launch(void* const* d_in, const int* in_sizes, int n_in,
                              void* d_out, int out_size, void* d_ws, size_t ws_size,
                              hipStream_t stream) {
  (void)in_sizes; (void)n_in; (void)out_size; (void)ws_size;
  const int*   inds     = (const int*)d_in[0];
  const float* emb      = (const float*)d_in[1];
  const float* w_ih_f   = (const float*)d_in[2];
  const float* w_hh_f   = (const float*)d_in[3];
  const float* b_ih_f   = (const float*)d_in[4];
  const float* b_hh_f   = (const float*)d_in[5];
  const float* w_ih_b   = (const float*)d_in[6];
  const float* w_hh_b   = (const float*)d_in[7];
  const float* b_ih_b   = (const float*)d_in[8];
  const float* b_hh_b   = (const float*)d_in[9];
  const float* w_local  = (const float*)d_in[10];
  const float* b_local  = (const float*)d_in[11];
  const float* w_global = (const float*)d_in[12];
  const float* b_global = (const float*)d_in[13];
  const float* w_common = (const float*)d_in[14];
  const float* b_common = (const float*)d_in[15];
  float* out = (float*)d_out;

  // bump allocator over workspace (256B aligned)
  size_t off = 0;
  auto alloc = [&](size_t bytes) -> void* {
    off = (off + 255) & ~(size_t)255;
    void* p = (char*)d_ws + off;
    off += bytes;
    return p;
  };
  unsigned short* x_bf  = (unsigned short*)alloc((size_t)NB * NT * NEP * 2);
  unsigned short* wihfb = (unsigned short*)alloc((size_t)NG * NEP * 2);
  unsigned short* wihbb = (unsigned short*)alloc((size_t)NG * NEP * 2);
  unsigned short* whhfb = (unsigned short*)alloc((size_t)NG * NH * 2);
  unsigned short* whhbb = (unsigned short*)alloc((size_t)NG * NH * 2);
  unsigned short* wlocb = (unsigned short*)alloc((size_t)ND * ND * 2);
  unsigned short* wglob = (unsigned short*)alloc((size_t)ND * ND * 2);
  float*          gi    = (float*)alloc((size_t)NB * NG * 4);
  float*          gh    = (float*)alloc((size_t)NB * NG * 4);
  float*          hf_f  = (float*)alloc((size_t)NB * NH * 4);
  float*          hf_b  = (float*)alloc((size_t)NB * NH * 4);
  unsigned short* hb_f  = (unsigned short*)alloc((size_t)NB * NH * 2);
  unsigned short* hb_b  = (unsigned short*)alloc((size_t)NB * NH * 2);
  float*          yw_f  = (float*)alloc((size_t)NB * NT * ND * 4);
  unsigned short* yw_bf = (unsigned short*)alloc((size_t)NB * NT * ND * 2);
  float*          yp_f  = (float*)alloc((size_t)NB * ND * 4);
  unsigned short* yp_bf = (unsigned short*)alloc((size_t)NB * ND * 2);
  float*          gemb  = (float*)alloc((size_t)NB * ND * 4);
  float*          wts   = (float*)alloc((size_t)NB * NT * 4);

  const int TB = 256;
  auto blocks = [](size_t n, int tb) { return (unsigned)((n + tb - 1) / tb); };

  // 1. weight conversion (f32 -> bf16, K padded to multiple of 32)
  convert_pad_kernel<<<blocks((size_t)NG * NEP, TB), TB, 0, stream>>>(w_ih_f, wihfb, NG, NE, NEP);
  convert_pad_kernel<<<blocks((size_t)NG * NEP, TB), TB, 0, stream>>>(w_ih_b, wihbb, NG, NE, NEP);
  convert_pad_kernel<<<blocks((size_t)NG * NH, TB), TB, 0, stream>>>(w_hh_f, whhfb, NG, NH, NH);
  convert_pad_kernel<<<blocks((size_t)NG * NH, TB), TB, 0, stream>>>(w_hh_b, whhbb, NG, NH, NH);
  convert_pad_kernel<<<blocks((size_t)ND * ND, TB), TB, 0, stream>>>(w_local, wlocb, ND, ND, ND);
  convert_pad_kernel<<<blocks((size_t)ND * ND, TB), TB, 0, stream>>>(w_global, wglob, ND, ND, ND);

  // 2. embedding gather -> padded bf16 activations [B*T, 320]
  gather_embed_kernel<<<blocks((size_t)NB * NT * NEP, TB), TB, 0, stream>>>(inds, emb, x_bf);

  // 3. zero hidden states
  fill_f32_kernel<<<blocks((size_t)NB * NH, TB), TB, 0, stream>>>(hf_f, 0.f, (size_t)NB * NH);
  fill_f32_kernel<<<blocks((size_t)NB * NH, TB), TB, 0, stream>>>(hf_b, 0.f, (size_t)NB * NH);
  fill_u16_kernel<<<blocks((size_t)NB * NH, TB), TB, 0, stream>>>(hb_f, 0, (size_t)NB * NH);
  fill_u16_kernel<<<blocks((size_t)NB * NH, TB), TB, 0, stream>>>(hb_b, 0, (size_t)NB * NH);

  const dim3 blk(128);
  const dim3 grdG(NG / 64, NB / 64);   // gate GEMMs: N=1536, M=4096
  const dim3 grdD(ND / 64, NB / 64);   // g_emb GEMM: N=1024, M=4096
  const dim3 grdA(ND / 64, (NB * NT) / 64); // attention GEMM: M=B*T
  const int ldaX = NT * NEP;           // x rows for fixed t are strided by T*320

  // 4. GRU time loops (forward then backward direction)
  for (int t = 0; t < NT; ++t) {
    wmma_gemm_bf16_kernel<<<grdG, blk, 0, stream>>>(x_bf + (size_t)t * NEP, ldaX, wihfb, NEP,
                                                    gi, NG, NEP, nullptr, 0);
    wmma_gemm_bf16_kernel<<<grdG, blk, 0, stream>>>(hb_f, NH, whhfb, NH, gh, NG, NH, nullptr, 0);
    gru_gate_kernel<<<blocks((size_t)NB * NH, TB), TB, 0, stream>>>(gi, gh, b_ih_f, b_hh_f,
                                                                    hf_f, hb_f, yw_f, yw_bf, t, 0);
  }
  for (int t = NT - 1; t >= 0; --t) {
    wmma_gemm_bf16_kernel<<<grdG, blk, 0, stream>>>(x_bf + (size_t)t * NEP, ldaX, wihbb, NEP,
                                                    gi, NG, NEP, nullptr, 0);
    wmma_gemm_bf16_kernel<<<grdG, blk, 0, stream>>>(hb_b, NH, whhbb, NH, gh, NG, NH, nullptr, 0);
    gru_gate_kernel<<<blocks((size_t)NB * NH, TB), TB, 0, stream>>>(gi, gh, b_ih_b, b_hh_b,
                                                                    hf_b, hb_b, yw_f, yw_bf, t, NH);
  }

  // 5. masked max-pool over valid tokens
  maxpool_kernel<<<blocks((size_t)NB * ND, TB), TB, 0, stream>>>(inds, yw_f, yp_f, yp_bf);

  // 6. g_emb = tanh(y_pool @ w_global^T + b_global)  (fused bias+tanh epilogue)
  wmma_gemm_bf16_kernel<<<grdD, blk, 0, stream>>>(yp_bf, ND, wglob, ND, gemb, ND, ND, b_global, 1);

  // 7. attention logits: init with b_common, accumulate fused tanh*g*wc reduction
  fill_from_ptr_kernel<<<blocks((size_t)NB * NT, TB), TB, 0, stream>>>(wts, b_common, (size_t)NB * NT);
  attn_weights_kernel<<<grdA, blk, 0, stream>>>(yw_bf, wlocb, b_local, gemb, w_common, wts);

  // 8. softmax over T and weighted sum -> [B, 1, D]
  softmax_wsum_kernel<<<NB, 128, 0, stream>>>(wts, yw_f, out);
}
